// StructuredAttentionLayer_81853486727220
// MI455X (gfx1250) — matrix operand order repaired and checked by reading, not compiled, not run on hardware
//
#include <hip/hip_runtime.h>
#include <hip/hip_bf16.h>

typedef __attribute__((ext_vector_type(16))) __bf16       v16bf;
typedef __attribute__((ext_vector_type(8)))  float        v8f;
typedef __attribute__((ext_vector_type(4)))  unsigned int u32x4;
typedef int i32x4 __attribute__((vector_size(16)));   // matches builtin param type

union Frag16 { v16bf v; u32x4 q[2]; };

__device__ __forceinline__ unsigned short f2bf(float f) {
  unsigned int u = __float_as_uint(f);
  u += 0x7FFFu + ((u >> 16) & 1u);        // round-to-nearest-even
  return (unsigned short)(u >> 16);
}
__device__ __forceinline__ float bf2f(unsigned short h) {
  return __uint_as_float(((unsigned int)h) << 16);
}

// -------- CDNA5 async global->LDS copy (ASYNCcnt path), guarded -----------
#if __has_builtin(__builtin_amdgcn_global_load_async_to_lds_b128) && \
    __has_builtin(__builtin_amdgcn_s_wait_asynccnt)
#define HAS_ASYNC 1
#else
#define HAS_ASYNC 0
#endif

template<int OFF>   // byte offset applied to BOTH global and LDS addresses
__device__ __forceinline__ void async_cp16(const unsigned short* g, unsigned short* l) {
#if HAS_ASYNC
  __builtin_amdgcn_global_load_async_to_lds_b128(
      (__attribute__((address_space(1))) i32x4*)(unsigned short*)g,
      (__attribute__((address_space(3))) i32x4*)l, OFF, 0);
#else
  *(u32x4*)((char*)l + OFF) = *(const u32x4*)((const char*)g + OFF);
#endif
}
__device__ __forceinline__ void wait_async() {
#if HAS_ASYNC
  __builtin_amdgcn_s_wait_asynccnt(0);
#endif
}

// ---------------------------------------------------------------------------
// C = relu(A @ Wt^T + bias) stored bf16.  A:[M,K] bf16, Wt:[N,K] bf16.
// M%128==0, K%32==0, N%128==0.
// 256 threads = 8 waves; block tile 128x128; wave tile 32x64 (2x4 WMMA accs).
// Double-buffered LDS tiles filled with global_load_async_to_lds_b128.
// ---------------------------------------------------------------------------
#define LDS_STRIDE 40   // 32 bf16 + 8 pad -> 80B rows, conflict-free 16B reads

__global__ __launch_bounds__(256)
void gemm_bf16(const unsigned short* __restrict__ A,
               const unsigned short* __restrict__ Wt,
               const float* __restrict__ bias,
               unsigned short* __restrict__ C, int M, int K, int N)
{
  __shared__ unsigned short ldsA[2][128 * LDS_STRIDE];   // [m][k]
  __shared__ unsigned short ldsB[2][128 * LDS_STRIDE];   // [n][k]

  const int tid   = threadIdx.x;
  const int lane  = tid & 31;
  const int wid   = tid >> 5;
  const int waveM = wid & 3;          // 0..3 -> 32-row slices
  const int waveN = wid >> 2;         // 0..1 -> 64-col slices
  const int half  = lane >> 4;
  const int l15   = lane & 15;
  const int rowBase = blockIdx.y * 128;
  const int colBase = blockIdx.x * 128;

  v8f zero = {0.f,0.f,0.f,0.f,0.f,0.f,0.f,0.f};
  v8f acc[2][4];
  #pragma unroll
  for (int i = 0; i < 2; ++i)
    #pragma unroll
    for (int j = 0; j < 4; ++j) acc[i][j] = zero;

  // per-thread copy coordinates: 128 rows x 32 bf16 per tile,
  // each thread moves one row-half (2 x 16B) of A and of B per k-step.
  const int r  = tid >> 1;             // 0..127
  const int c0 = (tid & 1) * 16;       // {0,16} bf16 elems
  const unsigned short* gA0 = A  + (long)(rowBase + r) * K + c0;
  const unsigned short* gB0 = Wt + (long)(colBase + r) * K + c0;
  unsigned short* lA = &ldsA[0][r * LDS_STRIDE + c0];
  unsigned short* lB = &ldsB[0][r * LDS_STRIDE + c0];
  const int bufStep = 128 * LDS_STRIDE;

  auto issue = [&](int buf, int kt) {
    async_cp16<0 >(gA0 + kt, lA + buf * bufStep);
    async_cp16<16>(gA0 + kt, lA + buf * bufStep);
    async_cp16<0 >(gB0 + kt, lB + buf * bufStep);
    async_cp16<16>(gB0 + kt, lB + buf * bufStep);
  };

  int cur = 0;
  issue(0, 0);
  for (int kt = 0; kt < K; kt += 32) {
    wait_async();
    __syncthreads();
    if (kt + 32 < K) issue(cur ^ 1, kt + 32);

    // A fragment: lane(m=l15, half h) holds k in [8h,8h+7] ++ [16+8h,16+8h+7]
    Frag16 a[2];
    #pragma unroll
    for (int tm = 0; tm < 2; ++tm) {
      const unsigned short* pa =
          &ldsA[cur][(waveM * 32 + tm * 16 + l15) * LDS_STRIDE + half * 8];
      a[tm].q[0] = *(const u32x4*)pa;
      a[tm].q[1] = *(const u32x4*)(pa + 16);
    }
    // B fragments: lane(n=l15, half h) holds k in [8h,8h+7] ++ [16+8h,16+8h+7]
    Frag16 b[4];
    #pragma unroll
    for (int tn = 0; tn < 4; ++tn) {
      const unsigned short* pb =
          &ldsB[cur][(waveN * 64 + tn * 16 + l15) * LDS_STRIDE + half * 8];
      b[tn].q[0] = *(const u32x4*)pb;
      b[tn].q[1] = *(const u32x4*)(pb + 16);
    }
    #pragma unroll
    for (int tn = 0; tn < 4; ++tn)
      #pragma unroll
      for (int tm = 0; tm < 2; ++tm)
        acc[tm][tn] = __builtin_amdgcn_wmma_f32_16x16x32_bf16(
            false, a[tm].v, false, b[tn].v, (short)0, acc[tm][tn], false, false);
    cur ^= 1;
  }

  // epilogue: D lane layout -> row = r + 8*half, col = l15
  #pragma unroll
  for (int tm = 0; tm < 2; ++tm)
    #pragma unroll
    for (int tn = 0; tn < 4; ++tn) {
      int col  = colBase + waveN * 64 + tn * 16 + l15;
      int row0 = rowBase + waveM * 32 + tm * 16 + half * 8;
      float bb = bias[col];
      #pragma unroll
      for (int rr = 0; rr < 8; ++rr) {
        float v = acc[tm][tn][rr] + bb;
        v = v > 0.f ? v : 0.f;
        C[(long)(row0 + rr) * N + col] = f2bf(v);
      }
    }
}

// ---------------------------------------------------------------------------
// flat f32 -> bf16 convert (n % 4 == 0)
// ---------------------------------------------------------------------------
__global__ __launch_bounds__(256)
void cvt_f32_bf16(const float* __restrict__ in, unsigned short* __restrict__ out, int n)
{
  int i = (blockIdx.x * 256 + threadIdx.x) * 4;
  if (i >= n) return;
  float4 v = *(const float4*)(in + i);
  out[i + 0] = f2bf(v.x);
  out[i + 1] = f2bf(v.y);
  out[i + 2] = f2bf(v.z);
  out[i + 3] = f2bf(v.w);
}

// ---------------------------------------------------------------------------
// in: [K,N] f32  ->  out: [N,K] bf16   (K,N multiples of 32)
// ---------------------------------------------------------------------------
__global__ __launch_bounds__(256)
void transpose_cvt(const float* __restrict__ in, unsigned short* __restrict__ out,
                   int K, int N)
{
  __shared__ unsigned short t[32][34];
  int n0 = blockIdx.x * 32, k0 = blockIdx.y * 32;
  int c = threadIdx.x & 31, r0 = threadIdx.x >> 5;
  #pragma unroll
  for (int r = r0; r < 32; r += 8)
    t[r][c] = f2bf(in[(long)(k0 + r) * N + n0 + c]);
  __syncthreads();
  #pragma unroll
  for (int r = r0; r < 32; r += 8)
    out[(long)(n0 + r) * K + k0 + c] = t[c][r];
}

// ---------------------------------------------------------------------------
// unary[r, 0:2] = H2[r,:] @ W2[1024,2] + b2 ; one wave per row
// ---------------------------------------------------------------------------
__global__ __launch_bounds__(256)
void proj_unary(const unsigned short* __restrict__ H2, const float* __restrict__ W2,
                const float* __restrict__ b2, float* __restrict__ unary, int rowBase)
{
  int r    = blockIdx.x * 8 + (threadIdx.x >> 5);
  int lane = threadIdx.x & 31;
  const unsigned short* hrow = H2 + (long)r * 1024;
  float a0 = 0.f, a1 = 0.f;
  #pragma unroll 4
  for (int t = 0; t < 32; ++t) {
    int k = lane + t * 32;
    float h = bf2f(hrow[k]);
    a0 += h * W2[k * 2 + 0];
    a1 += h * W2[k * 2 + 1];
  }
  #pragma unroll
  for (int off = 16; off > 0; off >>= 1) {
    a0 += __shfl_xor(a0, off, 32);
    a1 += __shfl_xor(a1, off, 32);
  }
  if (lane == 0) {
    long o = (long)(rowBase + r) * 2;
    unary[o + 0] = a0 + b2[0];
    unary[o + 1] = a1 + b2[1];
  }
}

// ---------------------------------------------------------------------------
// relation encoder: b3[row,0:3] for row in [0, bsz*21)
// ---------------------------------------------------------------------------
__global__ __launch_bounds__(256)
void rel_encoder(const float* __restrict__ ctx, const float* __restrict__ W0,
                 const float* __restrict__ b0, const float* __restrict__ W1,
                 const float* __restrict__ b1, float* __restrict__ b3, int rows)
{
  __shared__ float sW0[5 * 64], sb0[64], sW1[64 * 3], sb1[3];
  for (int i = threadIdx.x; i < 320; i += 256) sW0[i] = W0[i];
  for (int i = threadIdx.x; i < 64;  i += 256) sb0[i] = b0[i];
  for (int i = threadIdx.x; i < 192; i += 256) sW1[i] = W1[i];
  if (threadIdx.x < 3) sb1[threadIdx.x] = b1[threadIdx.x];
  __syncthreads();
  int row = blockIdx.x * 256 + threadIdx.x;
  if (row >= rows) return;
  float x0 = ctx[row*5+0], x1 = ctx[row*5+1], x2 = ctx[row*5+2],
        x3 = ctx[row*5+3], x4 = ctx[row*5+4];
  float y0 = 0.f, y1 = 0.f, y2 = 0.f;
  #pragma unroll 8
  for (int h = 0; h < 64; ++h) {
    float a = sb0[h] + x0*sW0[h] + x1*sW0[64+h] + x2*sW0[128+h]
                     + x3*sW0[192+h] + x4*sW0[256+h];
    a = a > 0.f ? a : 0.f;
    y0 += a * sW1[h*3+0];
    y1 += a * sW1[h*3+1];
    y2 += a * sW1[h*3+2];
  }
  b3[row*3+0] = y0 + sb1[0];
  b3[row*3+1] = y1 + sb1[1];
  b3[row*3+2] = y2 + sb1[2];
}

// ---------------------------------------------------------------------------
// exact marginals over 128 joint states of the K7 factor graph
// ---------------------------------------------------------------------------
__global__ __launch_bounds__(128)
void marginals(const float* __restrict__ unary, const float* __restrict__ b3,
               float* __restrict__ out, int Btot, int bsz)
{
  int b = blockIdx.x * 128 + threadIdx.x;
  if (b >= Btot) return;
  int j = b % bsz;

  float u0[7], u1[7];
  #pragma unroll
  for (int i = 0; i < 7; ++i) {
    long o = ((long)b * 7 + i) * 2;
    u0[i] = unary[o]; u1[i] = unary[o + 1];
  }
  float pA[21], pB[21], pC[21];
  #pragma unroll
  for (int p = 0; p < 21; ++p) {
    long o = ((long)j * 21 + p) * 3;
    pA[p] = b3[o]; pB[p] = b3[o + 1]; pC[p] = b3[o + 2];
  }
  const int PI[21] = {0,0,0,0,0,0,1,1,1,1,1,2,2,2,2,3,3,3,4,4,5};
  const int PJ[21] = {1,2,3,4,5,6,2,3,4,5,6,3,4,5,6,4,5,6,5,6,6};

  auto joint = [&](int s) -> float {
    float t = 0.f;
    #pragma unroll
    for (int i = 0; i < 7; ++i) t += ((s >> i) & 1) ? u1[i] : u0[i];
    #pragma unroll
    for (int p = 0; p < 21; ++p) {
      int k = ((s >> PI[p]) & 1) + ((s >> PJ[p]) & 1);
      t += (k == 0) ? pA[p] : ((k == 1) ? pB[p] : pC[p]);
    }
    return t;
  };

  float mx = -1e30f;
  for (int s = 0; s < 128; ++s) mx = fmaxf(mx, joint(s));

  float s1[7], s0[7];
  #pragma unroll
  for (int i = 0; i < 7; ++i) { s1[i] = 0.f; s0[i] = 0.f; }
  for (int s = 0; s < 128; ++s) {
    float e = __expf(joint(s) - mx);
    #pragma unroll
    for (int i = 0; i < 7; ++i) {
      if ((s >> i) & 1) s1[i] += e; else s0[i] += e;
    }
  }
  #pragma unroll
  for (int i = 0; i < 7; ++i)
    out[(long)b * 7 + i] = __logf(s1[i]) - __logf(s0[i]);
}

// ---------------------------------------------------------------------------
extern "C" void kernel_launch(void* const* d_in, const int* in_sizes, int n_in,
                              void* d_out, int out_size, void* d_ws, size_t ws_size,
                              hipStream_t stream)
{
  (void)in_sizes; (void)n_in; (void)out_size; (void)ws_size;
  const float* input = (const float*)d_in[0];   // [4,2048,7,512]
  const float* ctx   = (const float*)d_in[1];   // [2048,21,5]
  const float* ffW0  = (const float*)d_in[2];   // [512,1024]
  const float* ffb0  = (const float*)d_in[3];
  const float* ffW1  = (const float*)d_in[4];   // [1024,1024]
  const float* ffb1  = (const float*)d_in[5];
  const float* ffW2  = (const float*)d_in[6];   // [1024,2]
  const float* ffb2  = (const float*)d_in[7];
  const float* reW0  = (const float*)d_in[8];   // [5,64]
  const float* reb0  = (const float*)d_in[9];
  const float* reW1  = (const float*)d_in[10];  // [64,3]
  const float* reb1  = (const float*)d_in[11];

  const int Nn = 4, bsz = 2048, D = 512, H = 1024;
  const int Btot  = Nn * bsz;        // 8192
  const int Mtot  = Btot * 7;        // 57344 rows through the FF
  const int CHUNK = 8192;            // rows per chunk (multiple of 128)
  const int nchunk = Mtot / CHUNK;   // 7

  // workspace layout (~44 MB)
  char* ws = (char*)d_ws;
  unsigned short* Abuf = (unsigned short*)ws; ws += (size_t)CHUNK * D * 2;  // 8 MB
  unsigned short* H1   = (unsigned short*)ws; ws += (size_t)CHUNK * H * 2;  // 16 MB
  unsigned short* H2   = (unsigned short*)ws; ws += (size_t)CHUNK * H * 2;  // 16 MB
  unsigned short* Wt0  = (unsigned short*)ws; ws += (size_t)H * D * 2;      // 1 MB
  unsigned short* Wt1  = (unsigned short*)ws; ws += (size_t)H * H * 2;      // 2 MB
  float* unary = (float*)ws;                  ws += (size_t)Mtot * 2 * sizeof(float);
  float* b3    = (float*)ws;

  // one-shot weight transpose+convert: W[K,N] f32 -> Wt[N,K] bf16
  transpose_cvt<<<dim3(H / 32, D / 32), 256, 0, stream>>>(ffW0, Wt0, D, H);
  transpose_cvt<<<dim3(H / 32, H / 32), 256, 0, stream>>>(ffW1, Wt1, H, H);

  // relation encoder (independent of FF chain)
  rel_encoder<<<(bsz * 21 + 255) / 256, 256, 0, stream>>>(
      ctx, reW0, reb0, reW1, reb1, b3, bsz * 21);

  for (int c = 0; c < nchunk; ++c) {
    const float* Achunk = input + (size_t)c * CHUNK * D;
    cvt_f32_bf16<<<(CHUNK * D / 4 + 255) / 256, 256, 0, stream>>>(Achunk, Abuf, CHUNK * D);
    dim3 grid(H / 128, CHUNK / 128);   // 8 x 64
    gemm_bf16<<<grid, 256, 0, stream>>>(Abuf, Wt0, ffb0, H1, CHUNK, D, H);
    gemm_bf16<<<grid, 256, 0, stream>>>(H1,   Wt1, ffb1, H2, CHUNK, H, H);
    proj_unary<<<CHUNK / 8, 256, 0, stream>>>(H2, ffW2, ffb2, unary, c * CHUNK);
  }

  marginals<<<(Btot + 127) / 128, 128, 0, stream>>>(unary, b3, (float*)d_out, Btot, bsz);
}
